// SparseBlock_Conv2d_BN_ReLU_14671608283677
// MI455X (gfx1250) — matrix operand order repaired and checked by reading, not compiled, not run on hardware
//
#include <hip/hip_runtime.h>

typedef __attribute__((ext_vector_type(16))) __bf16 v16bf;
typedef __attribute__((ext_vector_type(8)))  float  v8f;
typedef __attribute__((ext_vector_type(4)))  float  f4;   // native vector for NT builtins

#define H 256
#define W 256
#define C 128
#define COUT 128
#define BS_IN 18
#define BS_OUT 16
#define BSTR 16
#define KTOT 1152      // 9 * 128
#define KSTEPS 36      // KTOT / 32

__device__ __forceinline__ unsigned short f32_to_bf16(float f) {
    unsigned int u = __float_as_uint(f);
    unsigned int r = u + 0x7FFFu + ((u >> 16) & 1u);   // round-to-nearest-even
    return (unsigned short)(r >> 16);
}

// --- w (HWIO f32) -> wt[cout][k] bf16, k = (ky*3+kx)*128 + ci ---
__global__ void prep_weights(const float* __restrict__ w, unsigned short* __restrict__ wt) {
    int idx = blockIdx.x * blockDim.x + threadIdx.x;
    if (idx >= COUT * KTOT) return;
    int co  = idx / KTOT;
    int k   = idx - co * KTOT;
    int pos = k >> 7;        // (ky*3+kx)
    int ci  = k & 127;
    wt[idx] = f32_to_bf16(w[((size_t)pos * C + ci) * COUT + co]);
}

// --- fold conv-bias + BN into scale/shift: out = relu(acc*a + d) ---
__global__ void prep_bn(const float* __restrict__ b, const float* __restrict__ gamma,
                        const float* __restrict__ beta, const float* __restrict__ mean,
                        const float* __restrict__ var,
                        float* __restrict__ a_s, float* __restrict__ d_s) {
    int c = threadIdx.x;
    if (c < COUT) {
        float s = gamma[c] * rsqrtf(var[c] + 1e-3f);
        a_s[c] = s;
        d_s[c] = (b[c] - mean[c]) * s + beta[c];
    }
}

// --- y = x (non-temporal 128-bit copy; keeps weights/patches hot in L2) ---
__global__ void copy_x(const f4* __restrict__ src, f4* __restrict__ dst, int n4) {
    int i = blockIdx.x * blockDim.x + threadIdx.x;
    int stride = gridDim.x * blockDim.x;
    for (; i < n4; i += stride) {
        f4 v = __builtin_nontemporal_load(&src[i]);
        __builtin_nontemporal_store(v, &dst[i]);
    }
}

typedef union { uint4 q[2]; v16bf v; } tile_t;

// --- one workgroup (16 waves) per active block ---
// wave layout: 8 row-pairs x 2 cout-groups; each wave = 2 M-tiles x 4 N-tiles
__launch_bounds__(512, 1)
__global__ void conv_block(const float* __restrict__ x,
                           const unsigned short* __restrict__ wt,
                           const float* __restrict__ a_s,
                           const float* __restrict__ d_s,
                           const int* __restrict__ active,
                           float* __restrict__ y) {
    __shared__ __align__(16) unsigned short spatch[BS_IN * BS_IN * C];   // 82944 B bf16 patch

    const int act = active[blockIdx.x];
    const int bn  = act >> 8;       // / (16*16) blocks per image
    const int rem = act & 255;
    const int by  = rem >> 4;
    const int bx  = rem & 15;

    const int tid = threadIdx.x;

    // ---- stage 18x18x128 patch, f32 -> bf16, zero-pad bottom/right ----
    const int h0 = by * BSTR, w0 = bx * BSTR;
    const float* xb = x + (size_t)bn * H * W * C;
    for (int j = tid; j < BS_IN * BS_IN * (C / 4); j += 512) {
        int cell = j >> 5;                 // 0..323
        int c4   = (j & 31) << 2;          // channel base (multiple of 4)
        int d = cell / BS_IN, e = cell - d * BS_IN;
        int gh = h0 + d, gw = w0 + e;
        f4 v = {0.f, 0.f, 0.f, 0.f};
        if (gh < H && gw < W)
            v = *(const f4*)(xb + ((size_t)gh * W + gw) * C + c4);
        uint2 pk;
        pk.x = (unsigned)f32_to_bf16(v.x) | ((unsigned)f32_to_bf16(v.y) << 16);
        pk.y = (unsigned)f32_to_bf16(v.z) | ((unsigned)f32_to_bf16(v.w) << 16);
        *(uint2*)&spatch[cell * C + c4] = pk;
    }
    __syncthreads();

    const int wave = tid >> 5;
    const int lane = tid & 31;
    const int m    = lane & 15;     // A-row / B-column within tile
    const int hi   = lane >> 4;     // lane half
    const int rp   = wave >> 1;     // row pair 0..7
    const int ntg  = wave & 1;      // cout group (4 N-tiles each)
    const int oy0  = rp * 2;
    const int oy1  = oy0 + 1;
    const int sub  = hi << 4;       // A K half = hi*16 (contiguous, matches B)

    v8f acc[8];                     // [row*4 + nt]
    v8f zero = {};
    #pragma unroll
    for (int t = 0; t < 8; ++t) acc[t] = zero;

    // lane-resident base for B: column n = (ntg*4 + nt)*16 + m, K half = hi*16
    const unsigned short* wtl = wt + (size_t)(ntg * 64 + m) * KTOT + (hi << 4);

    // ---- preload the 4 B tiles for k-step 0 ----
    tile_t B[4];
    #pragma unroll
    for (int nt = 0; nt < 4; ++nt) {
        const unsigned short* bp = wtl + (size_t)nt * 16 * KTOT;
        B[nt].q[0] = *(const uint4*)(bp);
        B[nt].q[1] = *(const uint4*)(bp + 8);
    }

    // A double buffer (LDS), 2 rows
    auto loadA = [&](int s, int oy, tile_t& A) {
        int pos = s >> 2;                 // ky*3+kx
        int c0  = (s & 3) << 5;
        int ky  = pos / 3, kx = pos - ky * 3;
        const unsigned short* ap =
            &spatch[(((oy + ky) * BS_IN + kx) + m) * C + c0 + sub];
        A.q[0] = *(const uint4*)(ap);
        A.q[1] = *(const uint4*)(ap + 8);
    };

    tile_t Abuf[2][2];                    // [parity][row]
    loadA(0, oy0, Abuf[0][0]);
    loadA(0, oy1, Abuf[0][1]);

    for (int s = 0; s < KSTEPS; ++s) {
        tile_t* Ac = Abuf[s & 1];
        tile_t* An = Abuf[(s + 1) & 1];
        if (s + 1 < KSTEPS) {             // prefetch next A tiles from LDS
            loadA(s + 1, oy0, An[0]);
            loadA(s + 1, oy1, An[1]);
        }

        const int sn = (s + 1 < KSTEPS) ? (s + 1) : s;   // clamp (redundant last reload)
        const unsigned short* bnext = wtl + (size_t)sn * 32;

        #pragma unroll
        for (int nt = 0; nt < 4; ++nt) {
            acc[nt] = __builtin_amdgcn_wmma_f32_16x16x32_bf16(
                false, Ac[0].v, false, B[nt].v, (short)0, acc[nt], false, false);
            acc[4 + nt] = __builtin_amdgcn_wmma_f32_16x16x32_bf16(
                false, Ac[1].v, false, B[nt].v, (short)0, acc[4 + nt], false, false);
            // refill this tile for the next k-step right after its last use
            const unsigned short* bp = bnext + (size_t)nt * 16 * KTOT;
            B[nt].q[0] = *(const uint4*)(bp);
            B[nt].q[1] = *(const uint4*)(bp + 8);
        }
    }

    // ---- epilogue: BN + ReLU, scatter 2 block rows into y ----
    #pragma unroll
    for (int r = 0; r < 2; ++r) {
        const int gy = by * 16 + oy0 + r;
        float* ybase = y + (((size_t)bn * H + gy) * W + (size_t)bx * 16) * C;
        #pragma unroll
        for (int nt = 0; nt < 4; ++nt) {
            const int n  = (ntg * 4 + nt) * 16 + m;
            const float sa = a_s[n];
            const float sd = d_s[n];
            #pragma unroll
            for (int v = 0; v < 8; ++v) {
                const int ox = v + (hi << 3);    // C/D layout: M = vgpr + 8*laneHalf
                float q = fmaxf(acc[r * 4 + nt][v] * sa + sd, 0.f);
                ybase[(size_t)ox * C + n] = q;
            }
        }
    }
}

extern "C" void kernel_launch(void* const* d_in, const int* in_sizes, int n_in,
                              void* d_out, int out_size, void* d_ws, size_t ws_size,
                              hipStream_t stream) {
    const float* x     = (const float*)d_in[0];
    const float* w     = (const float*)d_in[1];
    const float* b     = (const float*)d_in[2];
    const float* gamma = (const float*)d_in[3];
    const float* beta  = (const float*)d_in[4];
    const float* mean  = (const float*)d_in[5];
    const float* var   = (const float*)d_in[6];
    const int* active  = (const int*)d_in[7];
    float* y = (float*)d_out;

    // workspace: bf16 transposed weights (294912 B) + BN scale/shift
    unsigned short* wt = (unsigned short*)d_ws;
    float* a_s = (float*)((char*)d_ws + 294912);
    float* d_s = a_s + COUT;

    const int nActive = in_sizes[7];

    prep_weights<<<(COUT * KTOT + 255) / 256, 256, 0, stream>>>(w, wt);
    prep_bn<<<1, 128, 0, stream>>>(b, gamma, beta, mean, var, a_s, d_s);

    const int n4 = out_size / 4;
    copy_x<<<2048, 256, 0, stream>>>((const f4*)x, (f4*)y, n4);

    conv_block<<<nActive, 512, 0, stream>>>(x, wt, a_s, d_s, active, y);
}
